// EntityAttentionLayer_14757507629760
// MI455X (gfx1250) — matrix operand order repaired and checked by reading, not compiled, not run on hardware
//
#include <hip/hip_runtime.h>

// ---------------------------------------------------------------------------
// Fused entity-attention for MI455X (gfx1250, wave32, WMMA bf16->f32).
// Compute-bound (~550 FLOP/byte) -> V_WMMA_F32_16X16X32_BF16 everywhere.
// v2: 2-wide N-tile register blocking to halve LDS (ds_load) traffic per WMMA;
//     hoisted loop-invariant A-fragments in the attention GEMMs.
// ---------------------------------------------------------------------------

#define BS      2048
#define NE      64
#define NQ      16
#define IN_DIM  512
#define EMBED   512
#define OUT_DIM 512
#define NH      8
#define HD      64

typedef __bf16 bf_t;
typedef bf_t  v8bf  __attribute__((ext_vector_type(8)));
typedef bf_t  v16bf __attribute__((ext_vector_type(16)));
typedef float v8f   __attribute__((ext_vector_type(8)));

// ---- LDS layout (units: unsigned short / bf16) ----------------------------
#define SP    520                    // 512-col rows, +8 pad (16B-aligned, bank-spread)
#define SVP   72                     // 64-col rows, +8 pad
#define OFF_ATTN 0                   // [NH][NQ][SVP]      = 9216
#define OFF_O    (NH * NQ * SVP)     // [NQ][SP]           = 8320  (attn+O alias E)
#define OFF_E    0                   // [NE][SP]           = 33280 (dead before attn/O)
#define OFF_Q    (NE * SP)           // 33280
#define OFF_K    (OFF_Q + NQ * SP)   // 41600
#define OFF_VT   (OFF_K + NE * SP)   // 74880  (V stored transposed: [EMBED][SVP])
#define SMEM_SH  (OFF_VT + EMBED * SVP) // 111744 shorts = 223488 B  (< 320KB WGP LDS)

__device__ __forceinline__ unsigned short f2bf(float f) {
  unsigned u = __float_as_uint(f);
  u += 0x7FFFu + ((u >> 16) & 1u);          // round-to-nearest-even
  return (unsigned short)(u >> 16);
}

__device__ __forceinline__ v8f zero8() {
  v8f z = {0.f, 0.f, 0.f, 0.f, 0.f, 0.f, 0.f, 0.f};
  return z;
}

// A fragment (16x32 bf16): lane holds K = koffA..koffA+7 and +16..+23 (two 16B chunks)
__device__ __forceinline__ v16bf frag_a(const unsigned short* p) {
  v8bf lo = *(const v8bf*)(p);
  v8bf hi = *(const v8bf*)(p + 16);
  return __builtin_shufflevector(lo, hi, 0,1,2,3,4,5,6,7,8,9,10,11,12,13,14,15);
}
// B fragment (32x16 bf16): lane holds K = koffB..koffB+7 and +16..+23 (two 16B chunks)
__device__ __forceinline__ v16bf frag_b(const unsigned short* p) {
  v8bf lo = *(const v8bf*)(p);
  v8bf hi = *(const v8bf*)(p + 16);
  return __builtin_shufflevector(lo, hi, 0,1,2,3,4,5,6,7,8,9,10,11,12,13,14,15);
}

__device__ __forceinline__ v8f wmma_bf16(v16bf a, v16bf b, v8f c) {
  return __builtin_amdgcn_wmma_f32_16x16x32_bf16(false, a, false, b, (short)0, c,
                                                 false, false);
}

// K=512 GEMM for MC M-tiles x TWO N-tiles; every A-fragment feeds 2 WMMAs.
// aBase: LDS (stride SP); wrow0/wrow1: global bf16 weight rows for the 2 tiles.
template <int MC>
__device__ __forceinline__ void gemm2(const unsigned short* aBase,
                                      const unsigned short* wrow0,
                                      const unsigned short* wrow1,
                                      int lane15, int koffA, int koffB,
                                      v8f* acc0, v8f* acc1) {
#pragma unroll
  for (int m = 0; m < MC; ++m) { acc0[m] = zero8(); acc1[m] = zero8(); }
#pragma unroll 4
  for (int kt = 0; kt < 16; ++kt) {
    const int k0 = kt << 5;
    v16bf b0 = frag_b(wrow0 + k0 + koffB);
    v16bf b1 = frag_b(wrow1 + k0 + koffB);
#pragma unroll
    for (int m = 0; m < MC; ++m) {
      v16bf af = frag_a(&aBase[(m * 16 + lane15) * SP + k0 + koffA]);
      acc0[m] = wmma_bf16(af, b0, acc0[m]);
      acc1[m] = wmma_bf16(af, b1, acc1[m]);
    }
  }
}

// ---------------------------------------------------------------------------
__global__ void prep_bf16(const float* __restrict__ W_in,
                          const float* __restrict__ W_out,
                          unsigned short* __restrict__ ws) {
  const int NWIN = 3 * EMBED * IN_DIM;        // 786432
  const int NTOT = NWIN + OUT_DIM * EMBED;    // 1048576
  for (int idx = blockIdx.x * blockDim.x + threadIdx.x; idx < NTOT;
       idx += gridDim.x * blockDim.x) {
    float v = (idx < NWIN) ? W_in[idx] : W_out[idx - NWIN];
    ws[idx] = f2bf(v);
  }
}

// ---------------------------------------------------------------------------
__global__ void __launch_bounds__(256)
entity_attention(const float* __restrict__ entities,
                 const unsigned char* __restrict__ pre_mask,   // [BS][NE][NE]
                 const unsigned char* __restrict__ post_mask,  // [BS][NQ]
                 const unsigned short* __restrict__ wqkv,      // bf16 [3E][IN]
                 const unsigned short* __restrict__ wout,      // bf16 [OUT][E]
                 const float* __restrict__ b_out,
                 float* __restrict__ out) {
  __shared__ __align__(16) unsigned short sm[SMEM_SH];

  const int b      = blockIdx.x;
  const int tid    = threadIdx.x;
  const int lane   = tid & 31;
  const int w      = tid >> 5;       // wave id 0..7
  const int lane15 = lane & 15;
  const int half   = lane >> 4;
  const int koffA  = half << 3;      // A-frag K offset within 32
  const int koffB  = half << 3;      // B-frag K offset within 32

  // ---- Phase A0: stage entities[b] (64x512 f32) -> LDS bf16 ---------------
  const float* Eg = entities + (size_t)b * NE * IN_DIM;
  for (int i = tid; i < NE * IN_DIM / 4; i += 256) {
    const float4 v = ((const float4*)Eg)[i];
    const int idx = i * 4;
    unsigned short* d = &sm[OFF_E + (idx >> 9) * SP + (idx & 511)];
    d[0] = f2bf(v.x); d[1] = f2bf(v.y); d[2] = f2bf(v.z); d[3] = f2bf(v.w);
  }
  __syncthreads();

  // ---- Phase A1: QKV = E @ W_in^T  (96 N-tiles, 8 waves, 2-wide blocking) --
  // Q range (n<512): only M-tile 0 needed (queries = first 16 entities).
  v8f a0[4], a1[4];
  for (int i0 = 0; i0 < 4; i0 += 2) {                // Q tile pairs (nt, nt+8)
    const int n0 = (w + (i0 << 3)) << 4;
    const int n1 = n0 + 128;
    const unsigned short* wr0 = wqkv + (size_t)(n0 + lane15) * IN_DIM;
    const unsigned short* wr1 = wqkv + (size_t)(n1 + lane15) * IN_DIM;
    gemm2<1>(&sm[OFF_E], wr0, wr1, lane15, koffA, koffB, a0, a1);
#pragma unroll
    for (int r = 0; r < 8; ++r) {                    // rows 0..15 = queries
      const int row = r + (half << 3);
      sm[OFF_Q + row * SP + n0 + lane15] = f2bf(a0[0][r]);
      sm[OFF_Q + row * SP + n1 + lane15] = f2bf(a1[0][r]);
    }
  }
  for (int i0 = 4; i0 < 12; i0 += 2) {               // K/V tile pairs
    const int n0 = (w + (i0 << 3)) << 4;             // 512..1023 (K) or 1024.. (V)
    const int n1 = n0 + 128;                         // same region as n0
    const unsigned short* wr0 = wqkv + (size_t)(n0 + lane15) * IN_DIM;
    const unsigned short* wr1 = wqkv + (size_t)(n1 + lane15) * IN_DIM;
    gemm2<4>(&sm[OFF_E], wr0, wr1, lane15, koffA, koffB, a0, a1);
#pragma unroll
    for (int p = 0; p < 2; ++p) {
      const v8f* acc = p ? a1 : a0;
      const int col = (p ? n1 : n0) + lane15;
#pragma unroll
      for (int m = 0; m < 4; ++m) {
#pragma unroll
        for (int r = 0; r < 8; ++r) {
          const int row = m * 16 + r + (half << 3);  // entity index 0..63
          const unsigned short v = f2bf(acc[m][r]);
          if (col < 2 * EMBED)                       // K: row-major [NE][E]
            sm[OFF_K + row * SP + (col - EMBED)] = v;
          else                                       // V: transposed [E][NE]
            sm[OFF_VT + (col - 2 * EMBED) * SVP + row] = v;
        }
      }
    }
  }
  __syncthreads();   // E region now dead -> attn/O may alias it

  // ---- Phase B: attention, one head per wave ------------------------------
  const int h = w;
  const unsigned char* pmb = pre_mask + (size_t)b * NE * NE;

  // logits = Q_h (16xHD) @ K_h^T; Q fragments are t-invariant -> hoist
  v16bf qa[2];
#pragma unroll
  for (int kt = 0; kt < 2; ++kt)
    qa[kt] = frag_a(&sm[OFF_Q + lane15 * SP + h * HD + kt * 32 + koffA]);

  v8f lg[4];
#pragma unroll
  for (int t = 0; t < 4; ++t) {                      // 4 key tiles of 16
    v8f a = zero8();
#pragma unroll
    for (int kt = 0; kt < 2; ++kt) {
      const int k0 = h * HD + kt * 32;
      v16bf kb = frag_b(&sm[OFF_K + (t * 16 + lane15) * SP + k0 + koffB]);
      a = wmma_bf16(qa[kt], kb, a);
    }
    lg[t] = a;
  }

  // mask + softmax per query row; row q spans 16 lanes x 4 tiles
  const float scale = 0.125f;                        // 1/sqrt(HD)
  const float NEG_INF = -__builtin_inff();
#pragma unroll
  for (int r = 0; r < 8; ++r) {
    const int q = r + (half << 3);
    float x[4];
    float mx = NEG_INF;
#pragma unroll
    for (int t = 0; t < 4; ++t) {
      const int key = t * 16 + lane15;
      float v = lg[t][r] * scale;
      if (pmb[q * NE + key]) v = NEG_INF;            // mask TRUE -> -inf
      x[t] = v;
      mx = fmaxf(mx, v);
    }
    for (int s = 8; s >= 1; s >>= 1) mx = fmaxf(mx, __shfl_xor(mx, s, 32));
    float sum = 0.f;
#pragma unroll
    for (int t = 0; t < 4; ++t) {
      const float e = (x[t] == NEG_INF) ? 0.f : __expf(x[t] - mx);
      x[t] = e;
      sum += e;
    }
    for (int s = 8; s >= 1; s >>= 1) sum += __shfl_xor(sum, s, 32);
    const float inv = (sum > 0.f) ? __builtin_amdgcn_rcpf(sum) : 0.f; // all-masked -> 0
#pragma unroll
    for (int t = 0; t < 4; ++t)
      sm[OFF_ATTN + h * (NQ * SVP) + q * SVP + t * 16 + lane15] = f2bf(x[t] * inv);
  }

  // attn (16x64) @ V_h (64x64); attn fragments are dt-invariant -> hoist
  v16bf aa[2];
#pragma unroll
  for (int kt = 0; kt < 2; ++kt)
    aa[kt] = frag_a(&sm[OFF_ATTN + h * (NQ * SVP) + lane15 * SVP + kt * 32 + koffA]);

#pragma unroll
  for (int dt = 0; dt < 4; ++dt) {
    const int d0 = h * HD + dt * 16;
    v8f a = zero8();
#pragma unroll
    for (int kt = 0; kt < 2; ++kt) {
      v16bf vb = frag_b(&sm[OFF_VT + (d0 + lane15) * SVP + kt * 32 + koffB]);
      a = wmma_bf16(aa[kt], vb, a);
    }
#pragma unroll
    for (int r = 0; r < 8; ++r) {
      const int q = r + (half << 3);
      sm[OFF_O + q * SP + d0 + lane15] = f2bf(a[r]);
    }
  }
  __syncthreads();

  // ---- Phase C: Y = O(16x512) @ W_out^T + b_out, post-mask, store f32 -----
  for (int i0 = 0; i0 < 4; i0 += 2) {                // 32 N-tiles, 2-wide pairs
    const int n0 = (w + (i0 << 3)) << 4;
    const int n1 = n0 + 128;
    const unsigned short* wr0 = wout + (size_t)(n0 + lane15) * EMBED;
    const unsigned short* wr1 = wout + (size_t)(n1 + lane15) * EMBED;
    gemm2<1>(&sm[OFF_O], wr0, wr1, lane15, koffA, koffB, a0, a1);
#pragma unroll
    for (int p = 0; p < 2; ++p) {
      const int o = (p ? n1 : n0) + lane15;
      const float bias = b_out[o];
      const v8f* acc = p ? a1 : a0;
      float* orow = out + (size_t)b * NQ * OUT_DIM + o;
#pragma unroll
      for (int r = 0; r < 8; ++r) {
        const int q = r + (half << 3);
        float v = acc[0][r] + bias;
        if (post_mask[b * NQ + q]) v = 0.f;
        orow[(size_t)q * OUT_DIM] = v;
      }
    }
  }
}

// ---------------------------------------------------------------------------
extern "C" void kernel_launch(void* const* d_in, const int* in_sizes, int n_in,
                              void* d_out, int out_size, void* d_ws, size_t ws_size,
                              hipStream_t stream) {
  (void)in_sizes; (void)n_in; (void)out_size; (void)ws_size;
  const float*         entities  = (const float*)d_in[0];
  const unsigned char* pre_mask  = (const unsigned char*)d_in[1];
  const unsigned char* post_mask = (const unsigned char*)d_in[2];
  const float*         W_in      = (const float*)d_in[3];
  const float*         W_out     = (const float*)d_in[4];
  const float*         b_out     = (const float*)d_in[5];

  unsigned short* ws   = (unsigned short*)d_ws;      // 2 MB bf16 weights
  unsigned short* wqkv = ws;                         // [1536][512]
  unsigned short* wout = ws + 3 * EMBED * IN_DIM;    // [512][512]

  prep_bf16<<<512, 256, 0, stream>>>(W_in, W_out, ws);
  entity_attention<<<BS, 256, 0, stream>>>(entities, pre_mask, post_mask,
                                           wqkv, wout, b_out, (float*)d_out);
}
